// Pointer_5334349381887
// MI455X (gfx1250) — matrix operand order repaired and checked by reading, not compiled, not run on hardware
//
#include <hip/hip_runtime.h>
#include <hip/hip_bf16.h>
#include <math.h>

#define B_ 32
#define L_ 2048
#define D_ 512
#define A_ 256
#define NEG_INF_ (-1e30f)

typedef __attribute__((ext_vector_type(16))) __bf16 v16bf;
typedef __attribute__((ext_vector_type(8)))  float  v8f;

union BF16Frag { v16bf v; __bf16 h[16]; };

__device__ __forceinline__ __bf16 tobf(float f) { return (__bf16)f; }

__device__ __forceinline__ float eluf(float x) {   // jax.nn.elu, alpha=1
  return x > 0.f ? x : expm1f(x);
}

// ---------------- W0 -> bf16 copy (row-major, L2-resident, done once) ----------------
__global__ __launch_bounds__(256)
void w0bf_kernel(const float* __restrict__ W0, __bf16* __restrict__ w0bf) {
  const int i = (blockIdx.x * 256 + threadIdx.x) * 4;   // A_*D_ = 131072 elems
  float4 f = *(const float4*)(W0 + i);
  w0bf[i + 0] = tobf(f.x);
  w0bf[i + 1] = tobf(f.y);
  w0bf[i + 2] = tobf(f.z);
  w0bf[i + 3] = tobf(f.w);
}

// ---------------- p_st = elu(state @ W1^T) : [B, A] ----------------
__global__ __launch_bounds__(A_)
void pst_kernel(const float* __restrict__ state, const float* __restrict__ W1,
                float* __restrict__ pst) {
  __shared__ float st[D_];
  const int b = blockIdx.x;
  for (int i = threadIdx.x; i < D_; i += blockDim.x) st[i] = state[b * D_ + i];
  __syncthreads();
  const int a = threadIdx.x;                 // 256 threads, one output each
  const float* w = W1 + (size_t)a * D_;
  float acc = 0.f;
  #pragma unroll 8
  for (int d = 0; d < D_; ++d) acc += st[d] * w[d];
  pst[b * A_ + a] = eluf(acc);
}

// -------- fused pointer: logits[b,l] = mask ? elu(W2 . tanh(elu(ctx@W0^T)+pst)) : -1e30 --------
// Block = 256 threads = 8 waves. Wave owns a 16-row l-tile and all 256 A-columns
// as 16 WMMA f32 accumulators. Per 32-wide k-step the bf16 W0 slab (16 KB) is
// staged global->LDS with async-to-LDS DMA (ASYNCcnt), then each wave issues
// 16 v_wmma_f32_16x16x32_bf16 against an A fragment converted on the fly.
__global__ __launch_bounds__(256, 1)
void pointer_logits_kernel(const float* __restrict__ context,
                           const unsigned char* __restrict__ mask,
                           const __bf16* __restrict__ w0bf,
                           const float* __restrict__ W2,
                           const float* __restrict__ pst,
                           float* __restrict__ out) {
  __shared__ __bf16 lds_w0[A_ * 32];           // 16 KB: W0[:, k0:k0+32] bf16
  __shared__ float lds_pst[A_];
  __shared__ float lds_w2[A_];

  const int b     = blockIdx.x >> 4;           // 16 l-blocks per batch
  const int lblk  = blockIdx.x & 15;
  const int wave  = threadIdx.x >> 5;
  const int lane  = threadIdx.x & 31;
  const int half  = lane >> 4;                 // wave32 lane halves
  const int mlane = lane & 15;
  const int l0    = lblk * 128 + wave * 16;

  lds_pst[threadIdx.x] = pst[b * A_ + threadIdx.x];
  lds_w2[threadIdx.x]  = W2[threadIdx.x];

  // LDS byte offset of the staging buffer for async DMA (generic ptr low 32 bits)
  const unsigned lds_dst = (unsigned)(size_t)(&lds_w0[0]) + threadIdx.x * 64u;

  v8f acc[16];
  #pragma unroll
  for (int n = 0; n < 16; ++n)
    #pragma unroll
    for (int v = 0; v < 8; ++v) acc[n][v] = 0.f;

  // A-matrix row for this lane (16-bit A 16x32 layout: both halves carry M=mlane)
  const float* arow = context + (((size_t)b * L_) + l0 + mlane) * D_;

  for (int ks = 0; ks < 16; ++ks) {            // K = 512 in steps of 32
    const int k0 = ks * 32;
    __syncthreads();                           // previous slab fully consumed
    {
      // Thread t stages row a=t of the slab: 64 bytes = 4 x b128 async copies.
      // INST_OFFSET is added to both LDS and global addresses (ISA 10.7),
      // so offset:j*16 advances both sides in lockstep.
      const unsigned voff = threadIdx.x * (D_ * 2u) + (unsigned)k0 * 2u;
      asm volatile("global_load_async_to_lds_b128 %0, %1, %2"
                   :: "v"(lds_dst), "v"(voff), "s"(w0bf) : "memory");
      asm volatile("global_load_async_to_lds_b128 %0, %1, %2 offset:16"
                   :: "v"(lds_dst), "v"(voff), "s"(w0bf) : "memory");
      asm volatile("global_load_async_to_lds_b128 %0, %1, %2 offset:32"
                   :: "v"(lds_dst), "v"(voff), "s"(w0bf) : "memory");
      asm volatile("global_load_async_to_lds_b128 %0, %1, %2 offset:48"
                   :: "v"(lds_dst), "v"(voff), "s"(w0bf) : "memory");
      asm volatile("s_wait_asynccnt 0x0" ::: "memory");
    }
    __syncthreads();                           // slab visible to all waves

    // Build A fragment in contiguous K order (matches B):
    // VGPR v (pair j=2v,2v+1): K = half*16 + 2v + (j&1)
    BF16Frag afrag;
    #pragma unroll
    for (int v = 0; v < 8; ++v) {
      const int kb = half * 16 + v * 2;
      float2 f = *(const float2*)(arow + k0 + kb);
      afrag.h[2 * v]     = tobf(f.x);
      afrag.h[2 * v + 1] = tobf(f.y);
    }

    // B fragment: lane holds column n, halves split K (0..15 / 16..31)
    #pragma unroll
    for (int n = 0; n < 16; ++n) {
      const v16bf bfrag =
          *(const v16bf*)(lds_w0 + (n * 16 + mlane) * 32 + half * 16);
      acc[n] = __builtin_amdgcn_wmma_f32_16x16x32_bf16(
          false, afrag.v, false, bfrag, (short)0, acc[n], false, false);
    }
  }
  __syncthreads();

  // Epilogue: s_j partial = sum_n W2[col] * tanh(elu(pctx) + pst[col])
  float s[8];
  #pragma unroll
  for (int v = 0; v < 8; ++v) s[v] = 0.f;
  #pragma unroll
  for (int n = 0; n < 16; ++n) {
    const int col   = n * 16 + mlane;
    const float pa  = lds_pst[col];
    const float w2v = lds_w2[col];
    #pragma unroll
    for (int v = 0; v < 8; ++v)
      s[v] += w2v * tanhf(eluf(acc[n][v]) + pa);
  }
  // reduce the 16 N-lanes within each lane-half (C/D layout: row = v + 8*half)
  #pragma unroll
  for (int v = 0; v < 8; ++v) {
    float x = s[v];
    x += __shfl_xor(x, 1, 16);
    x += __shfl_xor(x, 2, 16);
    x += __shfl_xor(x, 4, 16);
    x += __shfl_xor(x, 8, 16);
    s[v] = x;
  }
  if (mlane == 0) {
    #pragma unroll
    for (int v = 0; v < 8; ++v) {
      const int row = l0 + half * 8 + v;
      const float sj = eluf(s[v]);
      out[(size_t)b * L_ + row] =
          mask[(size_t)b * L_ + row] ? sj : NEG_INF_;
    }
  }
}

// ---------------- softmax over L + c = a^T context : [B, D] ----------------
__global__ __launch_bounds__(D_)
void softmax_wsum_kernel(const float* __restrict__ logits,
                         const float* __restrict__ context,
                         float* __restrict__ cvec) {
  __shared__ float red[D_];
  __shared__ float aw[L_];
  const int b = blockIdx.x;
  const int t = threadIdx.x;
  const float* lg = logits + (size_t)b * L_;

  float m = -3.4e38f;
  for (int l = t; l < L_; l += D_) m = fmaxf(m, lg[l]);
  red[t] = m; __syncthreads();
  for (int s = D_ / 2; s > 0; s >>= 1) {
    if (t < s) red[t] = fmaxf(red[t], red[t + s]);
    __syncthreads();
  }
  const float mx = red[0];
  __syncthreads();

  float ssum = 0.f;
  for (int l = t; l < L_; l += D_) {
    const float e = __expf(lg[l] - mx);      // masked lanes -> 0
    aw[l] = e; ssum += e;
  }
  red[t] = ssum; __syncthreads();
  for (int s = D_ / 2; s > 0; s >>= 1) {
    if (t < s) red[t] += red[t + s];
    __syncthreads();
  }
  const float inv = 1.f / red[0];
  __syncthreads();

  const float* cb = context + (size_t)b * L_ * D_;
  float accd = 0.f;
  #pragma unroll 4
  for (int l = 0; l < L_; ++l) accd += aw[l] * cb[(size_t)l * D_ + t];
  cvec[b * D_ + t] = accd * inv;
}

// ---------------- GRU cell: state2 = GRUCell(c, state) ----------------
__global__ __launch_bounds__(D_)
void gru_kernel(const float* __restrict__ cvec, const float* __restrict__ state,
                const float* __restrict__ W_ih, const float* __restrict__ W_hh,
                const float* __restrict__ b_ih, const float* __restrict__ b_hh,
                float* __restrict__ state2) {
  __shared__ float cs[4 * D_];
  __shared__ float ss[4 * D_];
  const int b0 = blockIdx.x * 4;
  const int h  = threadIdx.x;                 // 512 threads, one h each
  for (int i = h; i < 4 * D_; i += D_) {
    cs[i] = cvec[b0 * D_ + i];
    ss[i] = state[b0 * D_ + i];
  }
  __syncthreads();

  const float* wr_i = W_ih + (size_t)h * D_;
  const float* wz_i = W_ih + (size_t)(D_ + h) * D_;
  const float* wn_i = W_ih + (size_t)(2 * D_ + h) * D_;
  const float* wr_h = W_hh + (size_t)h * D_;
  const float* wz_h = W_hh + (size_t)(D_ + h) * D_;
  const float* wn_h = W_hh + (size_t)(2 * D_ + h) * D_;

  for (int bb = 0; bb < 4; ++bb) {
    const float* cb = cs + bb * D_;
    const float* sb = ss + bb * D_;
    float ir = b_ih[h], iz = b_ih[D_ + h], in_ = b_ih[2 * D_ + h];
    float hr = b_hh[h], hz = b_hh[D_ + h], hn = b_hh[2 * D_ + h];
    #pragma unroll 4
    for (int d = 0; d < D_; ++d) {
      const float cd = cb[d], sd = sb[d];
      ir += cd * wr_i[d]; iz += cd * wz_i[d]; in_ += cd * wn_i[d];
      hr += sd * wr_h[d]; hz += sd * wz_h[d]; hn  += sd * wn_h[d];
    }
    const float r = 1.f / (1.f + __expf(-(ir + hr)));
    const float z = 1.f / (1.f + __expf(-(iz + hz)));
    const float n = tanhf(in_ + r * hn);
    state2[(b0 + bb) * D_ + h] = (1.f - z) * n + z * sb[h];
  }
}

extern "C" void kernel_launch(void* const* d_in, const int* in_sizes, int n_in,
                              void* d_out, int out_size, void* d_ws, size_t ws_size,
                              hipStream_t stream) {
  (void)in_sizes; (void)n_in; (void)out_size; (void)ws_size;
  const float*         context = (const float*)d_in[0];
  const unsigned char* mask    = (const unsigned char*)d_in[1];   // jnp bool = 1 byte
  const float*         state   = (const float*)d_in[2];
  const float*         W0      = (const float*)d_in[3];
  const float*         W1      = (const float*)d_in[4];
  const float*         W2      = (const float*)d_in[5];
  const float*         W_ih    = (const float*)d_in[6];
  const float*         W_hh    = (const float*)d_in[7];
  const float*         b_ih    = (const float*)d_in[8];
  const float*         b_hh    = (const float*)d_in[9];

  float* out_s = (float*)d_out;                // logits_s [B,L]
  float* out_e = out_s + (size_t)B_ * L_;      // logits_e [B,L]

  float* ws   = (float*)d_ws;
  float* pst0 = ws;                            // B*A
  float* pst1 = pst0 + B_ * A_;                // B*A
  float* cvec = pst1 + B_ * A_;                // B*D
  float* st2  = cvec + B_ * D_;                // B*D
  __bf16* w0bf = (__bf16*)(st2 + B_ * D_);     // A*D bf16 (256 KB)

  w0bf_kernel<<<(A_ * D_) / (256 * 4), 256, 0, stream>>>(W0, w0bf);
  pst_kernel<<<B_, A_, 0, stream>>>(state, W1, pst0);
  pointer_logits_kernel<<<B_ * 16, 256, 0, stream>>>(context, mask, w0bf, W2, pst0, out_s);
  softmax_wsum_kernel<<<B_, D_, 0, stream>>>(out_s, context, cvec);
  gru_kernel<<<B_ / 4, D_, 0, stream>>>(cvec, state, W_ih, W_hh, b_ih, b_hh, st2);
  pst_kernel<<<B_, A_, 0, stream>>>(st2, W1, pst1);
  pointer_logits_kernel<<<B_ * 16, 256, 0, stream>>>(context, mask, w0bf, W2, pst1, out_e);
}
